// TransitionNER_5342939316561
// MI455X (gfx1250) — compile-verified, matrix operand order checked by reading, not run on hardware
//
#include <hip/hip_runtime.h>

#define S_LEN 512
#define D_DIM 512
#define H_DIM 512
#define DA_DIM 128
#define A_NUM 3
#define NACT 768
#define MAXS 8
#define H4 2048
#define ENTK 1152   // 2H + DA

typedef __attribute__((ext_vector_type(16))) __bf16 v16bf;
typedef __attribute__((ext_vector_type(8)))  float  v8f;
typedef unsigned int v4u __attribute__((ext_vector_type(4)));
typedef int          v8i __attribute__((ext_vector_type(8)));
typedef int          v4i __attribute__((ext_vector_type(4)));

__device__ __forceinline__ float sigf(float x){ return 1.0f/(1.0f + __expf(-x)); }

// ---------------------------------------------------------------------------
// Tensor Data Mover: async 2D tile load Global -> LDS (CDNA5 ISA ch. 7/8).
// ---------------------------------------------------------------------------
__device__ __forceinline__ void tdm_load_2d(const float* gsrc, unsigned lds_off,
                                            unsigned w, unsigned rows, unsigned stride)
{
    unsigned long long ga = (unsigned long long)(uintptr_t)gsrc;
    v4u g0;
    g0[0] = 1u;                                        // count=1, user descriptor
    g0[1] = lds_off;                                   // lds_addr (bytes)
    g0[2] = (unsigned)(ga & 0xffffffffu);              // global_addr[31:0]
    g0[3] = (unsigned)((ga >> 32) & 0x01ffffffu)       // global_addr[56:32]
          | (2u << 30);                                // type=2 ("image")
    v8i g1;
    g1[0] = 0x00020000;                                // data_size=2 (4 bytes)
    g1[1] = (int)((w & 0xffffu) << 16);                // tensor_dim0[15:0]
    g1[2] = (int)(((w >> 16) & 0xffffu)
          | ((rows & 0xffffu) << 16));                 // tensor_dim1[15:0]
    g1[3] = (int)(((rows >> 16) & 0xffffu)
          | ((w & 0xffffu) << 16));                    // tile_dim0
    g1[4] = (int)(rows & 0xffffu);                     // tile_dim1
    g1[5] = (int)stride;                               // tensor_dim0_stride[31:0]
    g1[6] = 0; g1[7] = 0;
    v4i g2 = {0,0,0,0}, g3 = {0,0,0,0};
#if __has_include(<hip/amd_detail/amd_gfx1250_TDM.h>)
    v8i g4 = {0,0,0,0,0,0,0,0};
    __builtin_amdgcn_tensor_load_to_lds(g0, g1, g2, g3, g4, 0);
#else
    __builtin_amdgcn_tensor_load_to_lds(g0, g1, g2, g3, 0);
#endif
}

// ---------------------------------------------------------------------------
// 1) Integer scan over actions (control path is data-independent).
// ---------------------------------------------------------------------------
__global__ void scan_pointers_k(const int* __restrict__ sentence,
                                const int* __restrict__ actions,
                                int* __restrict__ iws)
{
    if (threadIdx.x != 0 || blockIdx.x != 0) return;
    int* tokb  = iws + 0*NACT; int* toks  = iws + 1*NACT;
    int* pspre = iws + 2*NACT; int* wslot = iws + 3*NACT;
    int* pbpre = iws + 4*NACT; int* actar = iws + 5*NACT;
    int* outnz = iws + 6*NACT; int* stknz = iws + 7*NACT;
    int tk[MAXS+1];
    #pragma unroll
    for (int i=0;i<=MAXS;++i) tk[i] = -1;
    int pb = S_LEN, ps = 0, po = 0;
    for (int t=0;t<NACT;++t){
        int a = actions[t];
        int bidx = S_LEN - pb; bidx = bidx < 0 ? 0 : (bidx > S_LEN-1 ? S_LEN-1 : bidx);
        int wb = sentence[bidx];
        int psc = ps < 0 ? 0 : (ps > MAXS ? MAXS : ps);
        int w = ps + 1; w = w < 0 ? 0 : (w > MAXS ? MAXS : w);
        int pbc = pb < 0 ? 0 : (pb > S_LEN ? S_LEN : pb);
        tokb[t]=wb; toks[t]=tk[psc]; pspre[t]=psc; wslot[t]=w;
        pbpre[t]=pbc; actar[t]=a; outnz[t]=(po>0); stknz[t]=(ps>0);
        if (a==0) tk[w]=wb;
        if (a==0||a==1) pb -= 1;
        if (a==0) ps += 1; else if (a==2) ps -= 1;
        if (a==1||a==2) po += 1;
    }
}

// ---------------------------------------------------------------------------
// 2) Gather per-step input rows.
// ---------------------------------------------------------------------------
__global__ void prep_inputs_k(const int* __restrict__ iws,
                              const int* __restrict__ sentence,
                              const float* __restrict__ word_embeds,
                              const float* __restrict__ relation_embeds,
                              float* __restrict__ Xb, float* __restrict__ Xs,
                              float* __restrict__ EntIn, float* __restrict__ WeRev)
{
    int t = blockIdx.x;
    int wb   = iws[0*NACT + t];
    int wsid = iws[1*NACT + t];
    int a    = iws[5*NACT + t];
    for (int d = threadIdx.x; d < D_DIM; d += blockDim.x){
        Xb[(size_t)t*D_DIM + d] = word_embeds[(size_t)wb*D_DIM + d];
        Xs[(size_t)t*D_DIM + d] = (wsid >= 0) ? word_embeds[(size_t)wsid*D_DIM + d] : 0.0f;
    }
    for (int j = threadIdx.x; j < DA_DIM; j += blockDim.x)
        EntIn[(size_t)t*ENTK + 2*H_DIM + j] = relation_embeds[a*DA_DIM + j];
    if (t < S_LEN){
        int sw = sentence[S_LEN - 1 - t];
        for (int d = threadIdx.x; d < D_DIM; d += blockDim.x)
            WeRev[(size_t)t*D_DIM + d] = word_embeds[(size_t)sw*D_DIM + d];
    }
}

// ---------------------------------------------------------------------------
// 3) WMMA GEMM: C[M x N] = A[M x K] @ W[N x K]^T  (optionally += )
// ---------------------------------------------------------------------------
__global__ __launch_bounds__(256) void gemm_bf16_wmma_k(
    const float* __restrict__ Amat, int lda,
    const float* __restrict__ Wmat, int ldw,
    float* __restrict__ Cmat, int ldc,
    int M, int N, int K, int accumulate)
{
    int tilesN = N >> 4;
    int tilesM = M >> 4;
    int wid = blockIdx.x * (blockDim.x >> 5) + (threadIdx.x >> 5);
    if (wid >= tilesM * tilesN) return;
    int tm = wid / tilesN, tn = wid % tilesN;
    int lane = threadIdx.x & 31;
    int half = lane >> 4;
    int r    = lane & 15;
    const float* Arow = Amat + (size_t)(tm*16 + r) * lda;
    const float* Wrow = Wmat + (size_t)(tn*16 + r) * ldw;
    float* Crow0 = Cmat + (size_t)(tm*16 + 8*half) * ldc + tn*16 + r;
    v8f acc;
    if (accumulate){
        #pragma unroll
        for (int v=0; v<8; ++v) acc[v] = Crow0[(size_t)v*ldc];
    } else {
        #pragma unroll
        for (int v=0; v<8; ++v) acc[v] = 0.0f;
    }
    for (int k0 = 0; k0 < K; k0 += 32){
        v16bf af, bfv;
        #pragma unroll
        for (int e=0; e<16; ++e){
            int k = (e < 8) ? (8*half + e) : (16 + 8*half + (e - 8));
            af[e]  = (__bf16)Arow[k0 + k];
            bfv[e] = (__bf16)Wrow[k0 + k];
        }
        acc = __builtin_amdgcn_wmma_f32_16x16x32_bf16(false, af, false, bfv,
                                                      (short)0, acc, false, false);
    }
    #pragma unroll
    for (int v=0; v<8; ++v) Crow0[(size_t)v*ldc] = acc[v];
}

// Fold LSTM biases (bih+bhh) into the precomputed Wx tables once.
__global__ void add_bias_rows_k(float* __restrict__ Mtx,
                                const float* __restrict__ bih,
                                const float* __restrict__ bhh)
{
    int t = blockIdx.x;
    for (int j = threadIdx.x; j < H4; j += blockDim.x)
        Mtx[(size_t)t*H4 + j] += bih[j] + bhh[j];
}

// One-time fp32 -> bf16 weight compression (halves scan L2 traffic).
__global__ void to_bf16_k(const float* __restrict__ src,
                          __bf16* __restrict__ dst, size_t n)
{
    size_t i = (size_t)blockIdx.x * blockDim.x + threadIdx.x;
    size_t stride = (size_t)gridDim.x * blockDim.x;
    for (; i < n; i += stride) dst[i] = (__bf16)src[i];
}

// ---------------------------------------------------------------------------
// 4) Constant recurrent terms for entity cells 3/4: kc = Whh@h0 + bih + bhh
// ---------------------------------------------------------------------------
__global__ void cell_const_k(const float* __restrict__ Whh5,
                             const float* __restrict__ bih5,
                             const float* __restrict__ bhh5,
                             const float* __restrict__ h0,
                             float* __restrict__ kc /* 2*H4: cell3 then cell4 */)
{
    int j = blockIdx.x * blockDim.x + threadIdx.x;
    if (j >= 2*H4) return;
    int cellv = 3 + (j >> 11);
    int jj = j & (H4-1);
    const float* wr = Whh5 + ((size_t)cellv*H4 + jj) * H_DIM;
    float s = bih5[cellv*H4 + jj] + bhh5[cellv*H4 + jj];
    for (int k=0;k<H_DIM;++k) s += wr[k]*h0[k];
    kc[j] = s;
}

// 5) Entity cell gates (h0,c0 constant): EntIn[:, colOff:colOff+512] = h-out
__global__ void entity_gate_k(const float* __restrict__ G,
                              const float* __restrict__ kc,
                              const float* __restrict__ c0,
                              float* __restrict__ EntIn, int colOff)
{
    int t = blockIdx.x;
    for (int j = threadIdx.x; j < H_DIM; j += blockDim.x){
        float zi = G[(size_t)t*H4 + j]           + kc[j];
        float zf = G[(size_t)t*H4 + H_DIM + j]   + kc[H_DIM + j];
        float zg = G[(size_t)t*H4 + 2*H_DIM + j] + kc[2*H_DIM + j];
        float zo = G[(size_t)t*H4 + 3*H_DIM + j] + kc[3*H_DIM + j];
        float ig = sigf(zi), fg = sigf(zf), gg = tanhf(zg), og = sigf(zo);
        float c2 = fg * c0[j] + ig * gg;
        EntIn[(size_t)t*ENTK + colOff + j] = og * tanhf(c2);
    }
}

__global__ void bias_tanh_k(float* __restrict__ Ent, const float* __restrict__ be)
{
    int t = blockIdx.x;
    for (int j = threadIdx.x; j < D_DIM; j += blockDim.x)
        Ent[(size_t)t*D_DIM + j] = tanhf(Ent[(size_t)t*D_DIM + j] + be[j]);
}

// 6) Action LSTM input projections: only 3 distinct actions (biases folded in)
__global__ void wxa_k(const float* __restrict__ Wih_a,
                      const float* __restrict__ action_embeds,
                      const float* __restrict__ bih_a,
                      const float* __restrict__ bhh_a,
                      float* __restrict__ WxA)
{
    int j = blockIdx.x * blockDim.x + threadIdx.x;
    if (j >= A_NUM*H4) return;
    int a = j / H4, jj = j % H4;
    const float* wr = Wih_a + (size_t)jj * DA_DIM;
    const float* x = action_embeds + a*DA_DIM;
    float s = bih_a[jj] + bhh_a[jj];
    for (int k=0;k<DA_DIM;++k) s += wr[k]*x[k];
    WxA[j] = s;
}

// ---------------------------------------------------------------------------
// 7) Buffer LSTM scan: bf16 Whh stream + TDM ping-pong row staging.
// ---------------------------------------------------------------------------
__global__ __launch_bounds__(1024) void buffer_scan_k(
    const float* __restrict__ WxBuf, const __bf16* __restrict__ WhhB,
    const float* __restrict__ h0, const float* __restrict__ c0,
    float* __restrict__ Hbuf)
{
    __shared__ float sh[H_DIM], sc[H_DIM], sz[H4];
    __shared__ float sWx[2][H4];
    int tid = threadIdx.x;
    bool wave0 = (tid >> 5) == 0;
    if (tid < H_DIM){ sh[tid]=h0[tid]; sc[tid]=c0[tid]; Hbuf[tid]=h0[tid]; }
    if (wave0)
        tdm_load_2d(WxBuf, (unsigned)(uintptr_t)&sWx[0][0], H4, 1, H4);
    __syncthreads();
    for (int t=0;t<S_LEN;++t){
        if (wave0){
            __builtin_amdgcn_s_wait_tensorcnt(0);
            if (t+1 < S_LEN)
                tdm_load_2d(WxBuf + (size_t)(t+1)*H4,
                            (unsigned)(uintptr_t)&sWx[(t+1)&1][0], H4, 1, H4);
        }
        __syncthreads();
        const float* wx = sWx[t & 1];
        #pragma unroll
        for (int rep=0;rep<2;++rep){
            int j = tid + rep*1024;
            float s = wx[j];
            const __bf16* wr = WhhB + (size_t)j * H_DIM;   // cell 0
            #pragma unroll 8
            for (int k=0;k<H_DIM;++k) s += (float)wr[k]*sh[k];
            sz[j] = s;
        }
        __syncthreads();
        if (tid < H_DIM){
            float ig=sigf(sz[tid]), fg=sigf(sz[H_DIM+tid]);
            float gg=tanhf(sz[2*H_DIM+tid]), og=sigf(sz[3*H_DIM+tid]);
            float c2 = fg*sc[tid] + ig*gg;
            float h2 = og*tanhf(c2);
            sc[tid]=c2; sh[tid]=h2;
            Hbuf[(size_t)(t+1)*H_DIM + tid] = h2;
        }
        __syncthreads();
    }
}

// 8) Action LSTM scan: WxA TDM-loaded once; bf16 Whh_a stream.
__global__ __launch_bounds__(1024) void action_scan_k(
    const int* __restrict__ iws, const float* __restrict__ WxA,
    const __bf16* __restrict__ WhaB,
    const float* __restrict__ h0, const float* __restrict__ c0,
    float* __restrict__ Haprev)
{
    __shared__ float sh[H_DIM], sc[H_DIM], sz[H4];
    __shared__ float sWxA[A_NUM*H4];
    int tid = threadIdx.x;
    bool wave0 = (tid >> 5) == 0;
    if (tid < H_DIM){ sh[tid]=h0[tid]; sc[tid]=c0[tid]; }
    if (wave0){
        tdm_load_2d(WxA, (unsigned)(uintptr_t)&sWxA[0], H4, A_NUM, H4);
        __builtin_amdgcn_s_wait_tensorcnt(0);
    }
    __syncthreads();
    const int* actar = iws + 5*NACT;
    for (int t=0;t<NACT;++t){
        int a = actar[t];
        if (tid < H_DIM) Haprev[(size_t)t*H_DIM + tid] = sh[tid];
        #pragma unroll
        for (int rep=0;rep<2;++rep){
            int j = tid + rep*1024;
            float s = sWxA[a*H4 + j];
            const __bf16* wr = WhaB + (size_t)j*H_DIM;
            #pragma unroll 8
            for (int k=0;k<H_DIM;++k) s += (float)wr[k]*sh[k];
            sz[j] = s;
        }
        __syncthreads();
        if (tid < H_DIM){
            float ig=sigf(sz[tid]), fg=sigf(sz[H_DIM+tid]);
            float gg=tanhf(sz[2*H_DIM+tid]), og=sigf(sz[3*H_DIM+tid]);
            float c2 = fg*sc[tid]+ig*gg;
            sc[tid]=c2; sh[tid]=og*tanhf(c2);
        }
        __syncthreads();
    }
}

// 9) Per-step classifier features from precomputed trajectories
__global__ void build_feats_k(const int* __restrict__ iws,
                              const float* __restrict__ Hbuf,
                              const float* __restrict__ Haprev,
                              const float* __restrict__ empty_emb,
                              float* __restrict__ Bfeat, float* __restrict__ Afeat)
{
    int t = blockIdx.x;
    int pb = iws[4*NACT + t];
    int bufnz = pb > 0;
    for (int j = threadIdx.x; j < H_DIM; j += blockDim.x){
        Bfeat[(size_t)t*H_DIM + j] = bufnz ? Hbuf[(size_t)pb*H_DIM + j] : empty_emb[j];
        Afeat[(size_t)t*H_DIM + j] = (t > 0) ? Haprev[(size_t)t*H_DIM + j] : empty_emb[j];
    }
}

// ---------------------------------------------------------------------------
// 10) Main serial scan: bf16 Whh/W1 streams + TDM-staged Wx rows.
// ---------------------------------------------------------------------------
__global__ __launch_bounds__(1024) void main_scan_k(
    const int* __restrict__ iws,
    const float* __restrict__ Gba,
    const float* __restrict__ WxShift, const float* __restrict__ WxOut,
    const float* __restrict__ WxRed,
    const __bf16* __restrict__ WhhB,   // cells 0..2 packed; use 1,2
    const __bf16* __restrict__ W1B,
    const float* __restrict__ b1,
    const float* __restrict__ W2, const float* __restrict__ b2,
    const float* __restrict__ h0, const float* __restrict__ c0,
    const float* __restrict__ empty_emb,
    float* __restrict__ outNll)
{
    __shared__ float sHs[(MAXS+1)*H_DIM];
    __shared__ float sCs[(MAXS+1)*H_DIM];
    __shared__ float sho[H_DIM], sco[H_DIM];
    __shared__ float su[H_DIM];
    __shared__ float sz[H4];
    __shared__ float sstk[H_DIM], sout[H_DIM], semp[H_DIM];
    __shared__ float slog[A_NUM];
    __shared__ float sWx[2][H4];
    int tid = threadIdx.x;
    bool wave0 = (tid >> 5) == 0;
    const int* pspre = iws + 2*NACT;
    const int* wslot = iws + 3*NACT;
    const int* actar = iws + 5*NACT;
    const int* outnz = iws + 6*NACT;
    const int* stknz = iws + 7*NACT;
    for (int j = tid; j < (MAXS+1)*H_DIM; j += 1024){ sHs[j]=0.f; sCs[j]=0.f; }
    __syncthreads();
    if (tid < H_DIM){
        sHs[tid]=h0[tid]; sCs[tid]=c0[tid];
        sho[tid]=h0[tid]; sco[tid]=c0[tid];
        semp[tid]=empty_emb[tid];
    }
    if (wave0){
        int a0 = actar[0];
        const float* s0 = (a0==0) ? WxShift : ((a0==1) ? WxOut : WxRed);
        tdm_load_2d(s0, (unsigned)(uintptr_t)&sWx[0][0], H4, 1, H4);
    }
    __syncthreads();
    for (int t=0;t<NACT;++t){
        int a = actar[t], psp = pspre[t], w = wslot[t];
        int snz = stknz[t], onz = outnz[t];
        if (wave0){
            __builtin_amdgcn_s_wait_tensorcnt(0);
            if (t+1 < NACT){
                int an = actar[t+1];
                const float* sn = (an==0) ? WxShift : ((an==1) ? WxOut : WxRed);
                tdm_load_2d(sn + (size_t)(t+1)*H4,
                            (unsigned)(uintptr_t)&sWx[(t+1)&1][0], H4, 1, H4);
            }
        }
        if (tid < H_DIM){
            sstk[tid] = snz ? sHs[psp*H_DIM + tid] : semp[tid];
            sout[tid] = onz ? sho[tid] : semp[tid];
        }
        __syncthreads();
        {   // u = tanh(Gba(buf+act parts) + b1 + W1[:,0:512]@stk + W1[:,1024:1536]@out)
            int j = tid & (H_DIM-1);
            int part = tid >> 9;
            const __bf16* wr = W1B + (size_t)j*H4 + (part ? 2*H_DIM : 0);
            const float* xv = part ? sout : sstk;
            float s = 0.f;
            #pragma unroll 8
            for (int k=0;k<H_DIM;++k) s += (float)wr[k]*xv[k];
            sz[tid] = s;
        }
        __syncthreads();
        if (tid < H_DIM)
            su[tid] = tanhf(Gba[(size_t)t*H_DIM + tid] + b1[tid] + sz[tid] + sz[H_DIM + tid]);
        __syncthreads();
        if (tid < A_NUM){
            const float* wr = W2 + (size_t)tid*H_DIM;
            float s = b2[tid];
            for (int k=0;k<H_DIM;++k) s += wr[k]*su[k];
            slog[tid] = s;
        }
        __syncthreads();
        if (tid == 0){
            float m = fmaxf(slog[0], fmaxf(slog[1], slog[2]));
            float e = __expf(slog[0]-m) + __expf(slog[1]-m) + __expf(slog[2]-m);
            outNll[t] = m + __logf(e) - slog[a];
        }
        // selected recurrent cell update (only one per step is live)
        const float* hsrc = (a == 0) ? (sHs + psp*H_DIM) : sho;
        int cellv = (a == 0) ? 1 : 2;
        const float* wx = sWx[t & 1];
        #pragma unroll
        for (int rep=0;rep<2;++rep){
            int j = tid + rep*1024;
            float s = wx[j];
            const __bf16* wr = WhhB + ((size_t)cellv*H4 + j)*H_DIM;
            #pragma unroll 8
            for (int k=0;k<H_DIM;++k) s += (float)wr[k]*hsrc[k];
            sz[j] = s;
        }
        __syncthreads();
        if (tid < H_DIM){
            float ig=sigf(sz[tid]), fg=sigf(sz[H_DIM+tid]);
            float gg=tanhf(sz[2*H_DIM+tid]), og=sigf(sz[3*H_DIM+tid]);
            if (a == 0){
                float c2 = fg*sCs[psp*H_DIM+tid] + ig*gg;
                sCs[w*H_DIM+tid]=c2; sHs[w*H_DIM+tid]=og*tanhf(c2);
            } else {
                float c2 = fg*sco[tid] + ig*gg;
                sco[tid]=c2; sho[tid]=og*tanhf(c2);
            }
        }
        __syncthreads();
    }
}

// ---------------------------------------------------------------------------
// Workspace layout (float units)
// ---------------------------------------------------------------------------
constexpr size_t O_INT   = 0;                        // 8192 floats reserved (ints)
constexpr size_t O_XB    = 8192;
constexpr size_t O_XS    = O_XB    + (size_t)NACT*D_DIM;
constexpr size_t O_WXSH  = O_XS    + (size_t)NACT*D_DIM;
constexpr size_t O_WXOUT = O_WXSH  + (size_t)NACT*H4;
constexpr size_t O_WXRED = O_WXOUT + (size_t)NACT*H4;
constexpr size_t O_G     = O_WXRED + (size_t)NACT*H4;     // shared temp (also WeRev)
constexpr size_t O_ENTIN = O_G     + (size_t)NACT*H4;
constexpr size_t O_ENT   = O_ENTIN + (size_t)NACT*ENTK;
constexpr size_t O_WXBUF = O_ENT   + (size_t)NACT*D_DIM;
constexpr size_t O_HBUF  = O_WXBUF + (size_t)S_LEN*H4;
constexpr size_t O_HA    = O_HBUF  + (size_t)(S_LEN+1)*H_DIM;
constexpr size_t O_BF    = O_HA    + (size_t)NACT*H_DIM;
constexpr size_t O_AF    = O_BF    + (size_t)NACT*H_DIM;
constexpr size_t O_GBA   = O_AF    + (size_t)NACT*H_DIM;
constexpr size_t O_KC    = O_GBA   + (size_t)NACT*H_DIM;
constexpr size_t O_WXA   = O_KC    + 2*H4;
// bf16 weight copies (sizes below in float units = bf16 elements / 2)
constexpr size_t O_WHHB  = O_WXA   + (size_t)A_NUM*H4;            // 3*H4*H_DIM bf16
constexpr size_t O_WHAB  = O_WHHB  + ((size_t)3*H4*H_DIM)/2;      // H4*H_DIM bf16
constexpr size_t O_W1B   = O_WHAB  + ((size_t)H4*H_DIM)/2;        // H_DIM*H4 bf16

static inline void launch_gemm(const float* A, int lda, const float* W, int ldw,
                               float* C, int ldc, int M, int N, int K, int acc,
                               hipStream_t stream)
{
    int tiles = (M/16) * (N/16);
    int blocks = (tiles + 7) / 8;           // 8 waves per 256-thread block
    gemm_bf16_wmma_k<<<blocks, 256, 0, stream>>>(A, lda, W, ldw, C, ldc, M, N, K, acc);
}

extern "C" void kernel_launch(void* const* d_in, const int* in_sizes, int n_in,
                              void* d_out, int out_size, void* d_ws, size_t ws_size,
                              hipStream_t stream) {
    const int*   sentence        = (const int*)  d_in[0];
    const int*   actions         = (const int*)  d_in[1];
    const float* word_embeds     = (const float*)d_in[2];
    const float* action_embeds   = (const float*)d_in[3];
    const float* relation_embeds = (const float*)d_in[4];
    const float* empty_emb       = (const float*)d_in[5];
    const float* h0              = (const float*)d_in[6];
    const float* c0              = (const float*)d_in[7];
    const float* Wih5            = (const float*)d_in[8];
    const float* Whh5            = (const float*)d_in[9];
    const float* bih5            = (const float*)d_in[10];
    const float* bhh5            = (const float*)d_in[11];
    const float* Wih_a           = (const float*)d_in[12];
    const float* Whh_a           = (const float*)d_in[13];
    const float* bih_a           = (const float*)d_in[14];
    const float* bhh_a           = (const float*)d_in[15];
    const float* W1              = (const float*)d_in[16];
    const float* b1              = (const float*)d_in[17];
    const float* W2              = (const float*)d_in[18];
    const float* b2              = (const float*)d_in[19];
    const float* We              = (const float*)d_in[20];
    const float* be              = (const float*)d_in[21];
    float* out = (float*)d_out;
    float* ws  = (float*)d_ws;
    int*   iws = (int*)d_ws;

    float* Xb    = ws + O_XB;    float* Xs    = ws + O_XS;
    float* WxSh  = ws + O_WXSH;  float* WxOut = ws + O_WXOUT;
    float* WxRed = ws + O_WXRED; float* G     = ws + O_G;
    float* EntIn = ws + O_ENTIN; float* Ent   = ws + O_ENT;
    float* WxBuf = ws + O_WXBUF; float* Hbuf  = ws + O_HBUF;
    float* Ha    = ws + O_HA;    float* Bfeat = ws + O_BF;
    float* Afeat = ws + O_AF;    float* Gba   = ws + O_GBA;
    float* kc    = ws + O_KC;    float* WxA   = ws + O_WXA;
    __bf16* WhhB = (__bf16*)(ws + O_WHHB);
    __bf16* WhaB = (__bf16*)(ws + O_WHAB);
    __bf16* W1B  = (__bf16*)(ws + O_W1B);

    // Phase 0: trajectories, gathers, constants, bf16 weight compression
    scan_pointers_k<<<1, 1, 0, stream>>>(sentence, actions, iws);
    to_bf16_k<<<2048, 256, 0, stream>>>(Whh5, WhhB, (size_t)3*H4*H_DIM);
    to_bf16_k<<<1024, 256, 0, stream>>>(Whh_a, WhaB, (size_t)H4*H_DIM);
    to_bf16_k<<<1024, 256, 0, stream>>>(W1, W1B, (size_t)H_DIM*H4);
    cell_const_k<<<16, 256, 0, stream>>>(Whh5, bih5, bhh5, h0, kc);
    wxa_k<<<24, 256, 0, stream>>>(Wih_a, action_embeds, bih_a, bhh_a, WxA);
    prep_inputs_k<<<NACT, 256, 0, stream>>>(iws, sentence, word_embeds,
                                            relation_embeds, Xb, Xs, EntIn, G);

    // Phase 1: buffer LSTM — WMMA input projections (+bias fold) then scan
    launch_gemm(G, D_DIM, Wih5 + 0*(size_t)H4*D_DIM, D_DIM, WxBuf, H4,
                S_LEN, H4, D_DIM, 0, stream);
    add_bias_rows_k<<<S_LEN, 256, 0, stream>>>(WxBuf, bih5 + 0*H4, bhh5 + 0*H4);
    buffer_scan_k<<<1, 1024, 0, stream>>>(WxBuf, WhhB, h0, c0, Hbuf);

    // Phase 2: all hoisted batched GEMMs (WMMA)
    launch_gemm(Xb, D_DIM, Wih5 + 1*(size_t)H4*D_DIM, D_DIM, WxSh,  H4, NACT, H4, D_DIM, 0, stream);
    add_bias_rows_k<<<NACT, 256, 0, stream>>>(WxSh, bih5 + 1*H4, bhh5 + 1*H4);
    launch_gemm(Xb, D_DIM, Wih5 + 2*(size_t)H4*D_DIM, D_DIM, WxOut, H4, NACT, H4, D_DIM, 0, stream);
    add_bias_rows_k<<<NACT, 256, 0, stream>>>(WxOut, bih5 + 2*H4, bhh5 + 2*H4);

    launch_gemm(Xs, D_DIM, Wih5 + 3*(size_t)H4*D_DIM, D_DIM, G, H4, NACT, H4, D_DIM, 0, stream);
    entity_gate_k<<<NACT, 256, 0, stream>>>(G, kc, c0, EntIn, 0);
    launch_gemm(Xs, D_DIM, Wih5 + 4*(size_t)H4*D_DIM, D_DIM, G, H4, NACT, H4, D_DIM, 0, stream);
    entity_gate_k<<<NACT, 256, 0, stream>>>(G, kc + H4, c0, EntIn, H_DIM);

    launch_gemm(EntIn, ENTK, We, ENTK, Ent, D_DIM, NACT, D_DIM, ENTK, 0, stream);
    bias_tanh_k<<<NACT, 256, 0, stream>>>(Ent, be);
    launch_gemm(Ent, D_DIM, Wih5 + 2*(size_t)H4*D_DIM, D_DIM, WxRed, H4, NACT, H4, D_DIM, 0, stream);
    add_bias_rows_k<<<NACT, 256, 0, stream>>>(WxRed, bih5 + 2*H4, bhh5 + 2*H4);

    // Phase 3: action LSTM (independent serial scan) + classifier buf/act parts
    action_scan_k<<<1, 1024, 0, stream>>>(iws, WxA, WhaB, h0, c0, Ha);
    build_feats_k<<<NACT, 256, 0, stream>>>(iws, Hbuf, Ha, empty_emb, Bfeat, Afeat);
    launch_gemm(Bfeat, H_DIM, W1 + H_DIM,   H4, Gba, H_DIM, NACT, H_DIM, H_DIM, 0, stream);
    launch_gemm(Afeat, H_DIM, W1 + 3*H_DIM, H4, Gba, H_DIM, NACT, H_DIM, H_DIM, 1, stream);

    // Phase 4: the remaining truly-serial main scan
    main_scan_k<<<1, 1024, 0, stream>>>(iws, Gba, WxSh, WxOut, WxRed,
                                        WhhB, W1B, b1, W2, b2,
                                        h0, c0, empty_emb, out);
}
